// LSTM_11132555231299
// MI455X (gfx1250) — matrix operand order repaired and checked by reading, not compiled, not run on hardware
//
#include <hip/hip_runtime.h>

// ---------------------------------------------------------------------------
// Bidirectional LayerNorm-LSTM for MI455X (gfx1250, wave32, WMMA bf16)
// B=32, T=512, I=512, H=512, G=4H=2048
// ---------------------------------------------------------------------------

#define B_  32
#define T_  512
#define I_  512
#define H_  512
#define G_  2048
#define EPS 1e-5f

typedef __attribute__((ext_vector_type(16))) __bf16 v16bf;
typedef __attribute__((ext_vector_type(8)))  __bf16 v8bf;
typedef __attribute__((ext_vector_type(8)))  float  v8f;

// fp32 -> bf16 round-to-nearest-even
__device__ __forceinline__ unsigned short f2bf(float f) {
    unsigned u = __float_as_uint(f);
    u += 0x7FFFu + ((u >> 16) & 1u);
    return (unsigned short)(u >> 16);
}

__device__ __forceinline__ float sigm(float x) {
    return 1.0f / (1.0f + __expf(-x));
}

// WMMA A-operand load, 16-bit A 16x32 layout (contiguous K, matching B):
//   lanes 0-15:  K=0-15
//   lanes 16-31: K=16-31
// row = lane & 15; per-lane K = 16*(lane>>4) + [0..15].
__device__ __forceinline__ v16bf load_wmma_a(const unsigned short* base,
                                             int row, int ld, int k0, int lane) {
    const unsigned short* p = base + (size_t)row * ld + k0 + 16 * (lane >> 4);
    v8bf lo = *reinterpret_cast<const v8bf*>(p);
    v8bf hi = *reinterpret_cast<const v8bf*>(p + 8);
    return __builtin_shufflevector(lo, hi, 0, 1, 2, 3, 4, 5, 6, 7,
                                           8, 9, 10, 11, 12, 13, 14, 15);
}

// WMMA B-operand load, 16-bit B 32x16 layout:
//   lane holds column N = lane&15, K = (lane>>4)*16 + [0..15] (contiguous).
// With weights stored (N, K) row-major this is one 32-byte contiguous load.
__device__ __forceinline__ v16bf load_wmma_b(const unsigned short* base,
                                             int nrow, int ld, int k0, int lane) {
    return *reinterpret_cast<const v16bf*>(base + (size_t)nrow * ld + k0 +
                                           16 * (lane >> 4));
}

// ------------------------------- convert -----------------------------------
__global__ void cvt_bf16_kernel(const float* __restrict__ in,
                                unsigned short* __restrict__ out, int n) {
    int i = blockIdx.x * blockDim.x + threadIdx.x;
    if (i < n) out[i] = f2bf(in[i]);
}

// ------------------------- zero h, c state ---------------------------------
__global__ void init_state_kernel(unsigned short* __restrict__ hb,
                                  float* __restrict__ c, int n) {
    int i = blockIdx.x * blockDim.x + threadIdx.x;
    if (i < n) { hb[i] = 0; c[i] = 0.0f; }
}

// ---------------------------------------------------------------------------
// gemm_x: xg_pre[(t*B+b), g] = x[b*T+t, :] . Wx[g, :] + bx[g]
// One wave computes one M-tile x 4 consecutive N-tiles (A reuse).
// grid = 4096 blocks x 256 thr (8 waves); waves = 1024 mtiles * 32 ngroups.
// ---------------------------------------------------------------------------
__global__ void gemm_x_kernel(const unsigned short* __restrict__ xb,
                              const unsigned short* __restrict__ wxb,
                              const float* __restrict__ bx,
                              float* __restrict__ xg) {
    const int lane = threadIdx.x & 31;
    const int wid  = threadIdx.x >> 5;
    const int w    = blockIdx.x * 8 + wid;        // [0, 32768)
    const int mtile  = w >> 5;                    // [0, 1024)
    const int ntile0 = (w & 31) * 4;              // [0, 128) step 4

    const int arow = mtile * 16 + (lane & 15);

    v8f acc[4];
    #pragma unroll
    for (int u = 0; u < 4; ++u) acc[u] = (v8f)(0.0f);

    for (int k0 = 0; k0 < I_; k0 += 32) {
        v16bf a = load_wmma_a(xb, arow, I_, k0, lane);
        #pragma unroll
        for (int u = 0; u < 4; ++u) {
            const int brow = (ntile0 + u) * 16 + (lane & 15);
            v16bf b = load_wmma_b(wxb, brow, I_, k0, lane);
            acc[u] = __builtin_amdgcn_wmma_f32_16x16x32_bf16(
                false, a, false, b, (short)0, acc[u], false, false);
        }
    }

    // C layout: VGPR r -> row = r + 8*(lane>>4), col = lane&15 (within tile)
    #pragma unroll
    for (int u = 0; u < 4; ++u) {
        const int col  = (ntile0 + u) * 16 + (lane & 15);
        const float bv = bx[col];
        #pragma unroll
        for (int r = 0; r < 8; ++r) {
            const int m  = mtile * 16 + r + 8 * (lane >> 4);  // row of x = b*T+t
            const int bb = m >> 9;                            // / T_
            const int tt = m & (T_ - 1);
            xg[((size_t)tt * B_ + bb) * G_ + col] = acc[u][r] + bv;
        }
    }
}

// ---------------------------------------------------------------------------
// ln_rows: in-place LayerNorm over G per row of xg (T*B rows), with g_x/b_x.
// ---------------------------------------------------------------------------
__global__ void ln_rows_kernel(float* __restrict__ xg,
                               const float* __restrict__ gx,
                               const float* __restrict__ bxn) {
    __shared__ float s_sum[256], s_sq[256];
    const int tid = threadIdx.x;
    float* row = xg + (size_t)blockIdx.x * G_;

    float v[8], sum = 0.0f, sq = 0.0f;
    #pragma unroll
    for (int k = 0; k < 8; ++k) {
        v[k] = row[tid + 256 * k];
        sum += v[k]; sq += v[k] * v[k];
    }
    s_sum[tid] = sum; s_sq[tid] = sq;
    __syncthreads();
    for (int off = 128; off > 0; off >>= 1) {
        if (tid < off) { s_sum[tid] += s_sum[tid + off]; s_sq[tid] += s_sq[tid + off]; }
        __syncthreads();
    }
    const float mu   = s_sum[0] * (1.0f / G_);
    const float var  = s_sq[0] * (1.0f / G_) - mu * mu;
    const float rstd = rsqrtf(var + EPS);
    #pragma unroll
    for (int k = 0; k < 8; ++k) {
        const int j = tid + 256 * k;
        row[j] = (v[k] - mu) * rstd * gx[j] + bxn[j];
    }
}

// ---------------------------------------------------------------------------
// gemm_h: preG[d, b, g] = h_bf16[d, b, :] . Wh[g, :] + bh[g]
// 512 waves: d(2) x mtile(2) x ntile(128). grid = 64 blocks x 256 thr.
// ---------------------------------------------------------------------------
__global__ void gemm_h_kernel(const unsigned short* __restrict__ hb,
                              const unsigned short* __restrict__ whb,
                              const float* __restrict__ bh,
                              float* __restrict__ preG) {
    const int lane = threadIdx.x & 31;
    const int wid  = threadIdx.x >> 5;
    const int w    = blockIdx.x * 8 + wid;   // [0, 512)
    const int d     = w >> 8;
    const int mtile = (w >> 7) & 1;
    const int ntile = w & 127;

    const unsigned short* A = hb + (size_t)d * B_ * H_;
    const int arow = mtile * 16 + (lane & 15);
    const int brow = ntile * 16 + (lane & 15);

    v8f acc = (v8f)(0.0f);
    for (int k0 = 0; k0 < H_; k0 += 32) {
        v16bf a = load_wmma_a(A,   arow, H_, k0, lane);
        v16bf b = load_wmma_b(whb, brow, H_, k0, lane);
        acc = __builtin_amdgcn_wmma_f32_16x16x32_bf16(
            false, a, false, b, (short)0, acc, false, false);
    }

    const int col  = ntile * 16 + (lane & 15);
    const float bv = bh[col];
    float* out = preG + (size_t)d * B_ * G_;
    #pragma unroll
    for (int r = 0; r < 8; ++r) {
        const int rowm = mtile * 16 + r + 8 * (lane >> 4);
        out[(size_t)rowm * G_ + col] = acc[r] + bv;
    }
}

// ---------------------------------------------------------------------------
// step_update: one block per (dir, batch) row.
//   gates = xg[t_in, b] + LN_G(preG[d,b]) ; c' = sig(f)*c + sig(i)*tanh(g)
//   h     = sig(o) * tanh(LN_H(c'))
// writes fp32 output and bf16 h for the next step's GEMM.
// grid = 64 blocks x 256 thr.
// ---------------------------------------------------------------------------
__global__ void step_update_kernel(int t,
                                   const float* __restrict__ xg,
                                   const float* __restrict__ preG,
                                   const float* __restrict__ gh,
                                   const float* __restrict__ bhn,
                                   const float* __restrict__ gc,
                                   const float* __restrict__ bcn,
                                   float* __restrict__ cbuf,
                                   unsigned short* __restrict__ hb,
                                   float* __restrict__ out) {
    __shared__ float s_sum[256], s_sq[256], c_sh[H_];
    const int tid = threadIdx.x;
    const int d   = blockIdx.x >> 5;
    const int b   = blockIdx.x & 31;
    const int t_in = (d == 0) ? t : (T_ - 1 - t);

    const float* xr = xg   + ((size_t)t_in * B_ + b) * G_;
    const float* pr = preG + ((size_t)d * B_ + b) * G_;
    float*       cr = cbuf + ((size_t)d * B_ + b) * H_;

    // ---- LN over the G=2048 recurrent pre-activations ----
    float pv[8], sum = 0.0f, sq = 0.0f;
    #pragma unroll
    for (int k = 0; k < 8; ++k) {
        pv[k] = pr[tid + 256 * k];
        sum += pv[k]; sq += pv[k] * pv[k];
    }
    s_sum[tid] = sum; s_sq[tid] = sq;
    __syncthreads();
    for (int off = 128; off > 0; off >>= 1) {
        if (tid < off) { s_sum[tid] += s_sum[tid + off]; s_sq[tid] += s_sq[tid + off]; }
        __syncthreads();
    }
    const float mu   = s_sum[0] * (1.0f / G_);
    const float rstd = rsqrtf(s_sq[0] * (1.0f / G_) - mu * mu + EPS);

    // ---- gates + cell update (2 hidden units per thread) ----
    float o_g[2], c_n[2];
    #pragma unroll
    for (int jj = 0; jj < 2; ++jj) {
        const int j = tid + 256 * jj;
        float gv[4];
        #pragma unroll
        for (int q = 0; q < 4; ++q) {
            const int idx = q * H_ + j;
            gv[q] = xr[idx] + (pr[idx] - mu) * rstd * gh[idx] + bhn[idx];
        }
        const float cn = sigm(gv[1]) * cr[j] + sigm(gv[0]) * tanhf(gv[2]);
        c_n[jj] = cn; o_g[jj] = gv[3];
        cr[j] = cn; c_sh[j] = cn;
    }
    __syncthreads();

    // ---- LN over c (H=512) ----
    float cs = c_n[0] + c_n[1];
    float cq = c_n[0] * c_n[0] + c_n[1] * c_n[1];
    s_sum[tid] = cs; s_sq[tid] = cq;
    __syncthreads();
    for (int off = 128; off > 0; off >>= 1) {
        if (tid < off) { s_sum[tid] += s_sum[tid + off]; s_sq[tid] += s_sq[tid + off]; }
        __syncthreads();
    }
    const float muc   = s_sum[0] * (1.0f / H_);
    const float rstdc = rsqrtf(s_sq[0] * (1.0f / H_) - muc * muc + EPS);

    // ---- h = sig(o) * tanh(LN(c)) ; emit output + bf16 h ----
    #pragma unroll
    for (int jj = 0; jj < 2; ++jj) {
        const int j = tid + 256 * jj;
        const float h = sigm(o_g[jj]) * tanhf((c_n[jj] - muc) * rstdc * gc[j] + bcn[j]);
        out[((size_t)b * T_ + t_in) * (2 * H_) + d * H_ + j] = h;
        hb[((size_t)d * B_ + b) * H_ + j] = f2bf(h);
    }
}

// ---------------------------------------------------------------------------
extern "C" void kernel_launch(void* const* d_in, const int* in_sizes, int n_in,
                              void* d_out, int out_size, void* d_ws, size_t ws_size,
                              hipStream_t stream) {
    const float* x   = (const float*)d_in[0];
    const float* Wx  = (const float*)d_in[1];
    const float* bx  = (const float*)d_in[2];
    const float* Wh  = (const float*)d_in[3];
    const float* bh  = (const float*)d_in[4];
    const float* g_x = (const float*)d_in[5];
    const float* b_x = (const float*)d_in[6];
    const float* g_h = (const float*)d_in[7];
    const float* b_h = (const float*)d_in[8];
    const float* g_c = (const float*)d_in[9];
    const float* b_c = (const float*)d_in[10];
    float* out = (float*)d_out;

    // workspace layout (256-B aligned sections)
    char* ws = (char*)d_ws;
    size_t off = 0;
    auto take = [&](size_t bytes) {
        void* p = ws + off;
        off += (bytes + 255) & ~(size_t)255;
        return p;
    };
    float*          xg   = (float*)take((size_t)T_ * B_ * G_ * 4);   // 128 MB
    unsigned short* xb   = (unsigned short*)take((size_t)B_ * T_ * I_ * 2);
    unsigned short* wxb  = (unsigned short*)take((size_t)G_ * I_ * 2);
    unsigned short* whb  = (unsigned short*)take((size_t)G_ * H_ * 2);
    float*          preG = (float*)take((size_t)2 * B_ * G_ * 4);
    unsigned short* hb   = (unsigned short*)take((size_t)2 * B_ * H_ * 2);
    float*          cbuf = (float*)take((size_t)2 * B_ * H_ * 4);
    (void)ws_size; (void)in_sizes; (void)n_in; (void)out_size;

    // 1) fp32 -> bf16 staging
    {
        int n = B_ * T_ * I_;
        cvt_bf16_kernel<<<(n + 255) / 256, 256, 0, stream>>>(x, xb, n);
        n = G_ * I_;
        cvt_bf16_kernel<<<(n + 255) / 256, 256, 0, stream>>>(Wx, wxb, n);
        n = G_ * H_;
        cvt_bf16_kernel<<<(n + 255) / 256, 256, 0, stream>>>(Wh, whb, n);
    }

    // 2) zero h, c
    {
        int n = 2 * B_ * H_;
        init_state_kernel<<<(n + 255) / 256, 256, 0, stream>>>(hb, cbuf, n);
    }

    // 3) input projection GEMM (WMMA) + bias, then row LayerNorm
    gemm_x_kernel<<<4096, 256, 0, stream>>>(xb, wxb, bx, xg);
    ln_rows_kernel<<<T_ * B_, 256, 0, stream>>>(xg, g_x, b_x);

    // 4) recurrence: both directions per launch pair
    for (int t = 0; t < T_; ++t) {
        gemm_h_kernel<<<64, 256, 0, stream>>>(hb, whb, bh, preG);
        step_update_kernel<<<64, 256, 0, stream>>>(t, xg, preG, g_h, b_h,
                                                   g_c, b_c, cbuf, hb, out);
    }
}
